// PointCloudNet_45208825757826
// MI455X (gfx1250) — matrix operand order repaired and checked by reading, not compiled, not run on hardware
//
#include <hip/hip_runtime.h>
#include <hip/hip_bf16.h>
#include <stdint.h>

typedef __attribute__((ext_vector_type(16))) _Float16 v16h;
typedef __attribute__((ext_vector_type(8)))  _Float16 v8h;
typedef __attribute__((ext_vector_type(8)))  float    v8f;

// =====================================================================
// Fused linear layer via V_WMMA_F32_16X16X32_F16:
//   Y = relu((X @ W^T + bias) * gamma + beta),  f32 accumulate
// X:  (M x Kpad) f16 row-major, zero-padded K.
// Wp: (N x Kpad) f16 row-major, zero-padded K (pre-packed from f32 W).
// Block = 128 threads = 4 waves; each wave owns a 16x16 tile; the 4
// waves cover 64 consecutive M rows and share the same B tile (L0 reuse).
// =====================================================================

// A fragment 16x32 (ISA 7.12.2): lane L -> M=L%16; VGPRs 0..3 hold
// K = {2v,2v+1} + 8*half, VGPRs 4..7 the same +16.  Two contiguous
// 8-half (16B) chunks per lane: [k0+8*half, +8) and [k0+16+8*half, +8).
__device__ __forceinline__ v16h load_a(const _Float16* __restrict__ row,
                                       int k0, int half) {
  const v8h lo = *(const v8h*)(row + k0 + (half << 3));
  const v8h hi = *(const v8h*)(row + k0 + 16 + (half << 3));
  v16h a;
#pragma unroll
  for (int i = 0; i < 8; ++i) { a[i] = lo[i]; a[8 + i] = hi[i]; }
  return a;
}

// B fragment 32x16: lane L -> N=L%16; same K order as A: two 8-half
// chunks per lane: [k0+8*half, +8) and [k0+16+8*half, +8).
__device__ __forceinline__ v16h load_b(const _Float16* __restrict__ wrow,
                                       int k0, int half) {
  const v8h lo = *(const v8h*)(wrow + k0 + (half << 3));
  const v8h hi = *(const v8h*)(wrow + k0 + 16 + (half << 3));
  v16h b;
#pragma unroll
  for (int i = 0; i < 8; ++i) { b[i] = lo[i]; b[8 + i] = hi[i]; }
  return b;
}

__global__ void linear_relu_wmma(const _Float16* __restrict__ X,
                                 const _Float16* __restrict__ Wp,
                                 const float* __restrict__ bias,
                                 const float* __restrict__ gamma,
                                 const float* __restrict__ beta,
                                 _Float16* __restrict__ Y,
                                 int Kpad, int N) {
  const int lane = threadIdx.x & 31;   // EXEC all ones per wave
  const int wid  = threadIdx.x >> 5;
  const int m0 = (blockIdx.x << 6) + (wid << 4);
  const int n0 = blockIdx.y << 4;
  const int half = lane >> 4;
  const _Float16* arow = X  + (size_t)(m0 + (lane & 15)) * Kpad;
  const _Float16* wrow = Wp + (size_t)(n0 + (lane & 15)) * Kpad;

  v8f c = {};
  int k0 = 0;
  for (; k0 + 64 <= Kpad; k0 += 64) {
    // locality 3 -> WGP scope: pull into all cache levels on miss
    __builtin_prefetch(arow + k0 + 64, 0, 3);   // global_prefetch_b8
    __builtin_prefetch(wrow + k0 + 64, 0, 3);
    const v16h a0 = load_a(arow, k0, half);
    const v16h b0 = load_b(wrow, k0, half);
    const v16h a1 = load_a(arow, k0 + 32, half);
    const v16h b1 = load_b(wrow, k0 + 32, half);
    c = __builtin_amdgcn_wmma_f32_16x16x32_f16(false, a0, false, b0, (short)0,
                                               c, false, false);
    c = __builtin_amdgcn_wmma_f32_16x16x32_f16(false, a1, false, b1, (short)0,
                                               c, false, false);
  }
  for (; k0 < Kpad; k0 += 32) {
    const v16h a0 = load_a(arow, k0, half);
    const v16h b0 = load_b(wrow, k0, half);
    c = __builtin_amdgcn_wmma_f32_16x16x32_f16(false, a0, false, b0, (short)0,
                                               c, false, false);
  }

  // C/D layout: VGPR v -> M = m0 + v + 8*half, N = n0 + lane%16
  const int n = n0 + (lane & 15);
  const float bb = bias[n], gg = gamma[n], be = beta[n];
#pragma unroll
  for (int v = 0; v < 8; ++v) {
    const int m = m0 + v + (half << 3);
    float val = (c[v] + bb) * gg + be;
    val = val > 0.0f ? val : 0.0f;
    Y[(size_t)m * N + n] = (_Float16)val;
  }
}

// =====================================================================
// One-time weight pre-pack: f32 (N x K) -> f16 (N x Kpad), zero padded.
// =====================================================================
__global__ void pack_w_f16(const float* __restrict__ W,
                           _Float16* __restrict__ Wp, int N, int K, int Kpad) {
  const int idx = blockIdx.x * blockDim.x + threadIdx.x;
  if (idx >= N * Kpad) return;
  const int n = idx / Kpad;
  const int k = idx - n * Kpad;
  Wp[idx] = (k < K) ? (_Float16)W[(size_t)n * K + k] : (_Float16)0.0f;
}

// =====================================================================
// Build padded f16 input (xyz channels of the pointcloud, K padded to 32)
// =====================================================================
__global__ void build_xyz_f16(const float* __restrict__ pc,  // N x 6
                              _Float16* __restrict__ X, int N, int Kpad) {
  const int i = blockIdx.x * blockDim.x + threadIdx.x;
  if (i >= N) return;
  _Float16* row = X + (size_t)i * Kpad;
  row[0] = (_Float16)pc[i * 6 + 0];
  row[1] = (_Float16)pc[i * 6 + 1];
  row[2] = (_Float16)pc[i * 6 + 2];
  for (int k = 3; k < Kpad; ++k) row[k] = (_Float16)0.0f;
}

// =====================================================================
// Column mean of final activations: out[n] = (1/M) * sum_m Y[m][n]
// =====================================================================
__global__ void colmean_kernel(const _Float16* __restrict__ Y, int M, int N,
                               float* __restrict__ out) {
  const int n = blockIdx.x * blockDim.x + threadIdx.x;
  if (n >= N) return;
  float s = 0.0f;
  for (int m = 0; m < M; ++m) s += (float)Y[(size_t)m * N + n];
  out[n] = s / (float)M;
}

// =====================================================================
// Farthest-point sampling. One block per batch; dists live in LDS.
// jax scan semantics: out[t] is carried 'far' BEFORE the update;
// argmax tie-break = lowest index.  Per-wave shuffle reduction + small
// LDS combine -> only 2 barriers per serial step.
// =====================================================================
__device__ __forceinline__ void argmax_shfl(float& v, int& i) {
#pragma unroll
  for (int o = 16; o > 0; o >>= 1) {
    const float ov = __shfl_down(v, o, 32);
    const int   oi = __shfl_down(i, o, 32);
    if (ov > v || (ov == v && oi < i)) { v = ov; i = oi; }
  }
}

__global__ void fps_kernel(const float* __restrict__ xyz, int N, int stride,
                           int* __restrict__ out, int npoint) {
  __shared__ float dists[4096];
  __shared__ float rmax[16];
  __shared__ int   ridx[16];
  __shared__ int   s_far;
  const int tid = threadIdx.x;
  const int nt = blockDim.x;
  const int lane = tid & 31;
  const int wid = tid >> 5;
  const int nw = nt >> 5;
  for (int j = tid; j < N; j += nt) dists[j] = 1.0e10f;
  if (tid == 0) s_far = 0;
  __syncthreads();
  for (int t = 0; t < npoint; ++t) {
    const int far = s_far;
    if (tid == 0) out[t] = far;
    const float cx = xyz[far * stride + 0];
    const float cy = xyz[far * stride + 1];
    const float cz = xyz[far * stride + 2];
    float bestv = -1.0f;
    int besti = N;
    for (int j = tid; j < N; j += nt) {
      const float dx = xyz[j * stride + 0] - cx;
      const float dy = xyz[j * stride + 1] - cy;
      const float dz = xyz[j * stride + 2] - cz;
      const float d = dx * dx + dy * dy + dz * dz;
      const float dm = fminf(dists[j], d);
      dists[j] = dm;
      if (dm > bestv || (dm == bestv && j < besti)) { bestv = dm; besti = j; }
    }
    argmax_shfl(bestv, besti);
    if (lane == 0) { rmax[wid] = bestv; ridx[wid] = besti; }
    __syncthreads();
    if (wid == 0) {
      float v = (lane < nw) ? rmax[lane] : -1.0f;
      int   i = (lane < nw) ? ridx[lane] : N;
      argmax_shfl(v, i);
      if (lane == 0) s_far = i;
    }
    __syncthreads();
  }
}

// gather selected centroids into a compact (S x 3) f32 buffer
__global__ void gather_xyz(const float* __restrict__ xyz, int stride,
                           const int* __restrict__ idx,
                           float* __restrict__ out, int S) {
  const int s = blockIdx.x * blockDim.x + threadIdx.x;
  if (s >= S) return;
  const int j = idx[s];
  out[s * 3 + 0] = xyz[j * stride + 0];
  out[s * 3 + 1] = xyz[j * stride + 1];
  out[s * 3 + 2] = xyz[j * stride + 2];
}

// =====================================================================
// RBF mean: out[k] = mean_i exp(-||x_i - c_k||^2 * exp(2*log_sigma_k))
// =====================================================================
__global__ void rbf_kernel(const float* __restrict__ xyz, int S,
                           const float* __restrict__ centres,
                           const float* __restrict__ logsig, int K,
                           float* __restrict__ out) {
  const int k = blockIdx.x * blockDim.x + threadIdx.x;
  if (k >= K) return;
  const float cx = centres[k * 3 + 0];
  const float cy = centres[k * 3 + 1];
  const float cz = centres[k * 3 + 2];
  const float a = __expf(2.0f * logsig[k]);
  float s = 0.0f;
  for (int i = 0; i < S; ++i) {
    const float dx = xyz[i * 3 + 0] - cx;
    const float dy = xyz[i * 3 + 1] - cy;
    const float dz = xyz[i * 3 + 2] - cz;
    const float d2 = dx * dx + dy * dy + dz * dz;
    s += expf(-d2 * a);
  }
  out[k] = s / (float)S;
}

// =====================================================================
// Host orchestration
// Input order (jax pytree, dict order):
//   [0]            pointcloud (8,4096,6)
//   [1..24]        global_params: 6 x (W, b, g, beta)
//   [25..72]       sa_params: 4 x 3 x (W, b, g, beta)   -- DEAD CODE (unused:
//                  SA feature MLPs never reach the output; FPS uses xyz only)
//   [73..78]       rbf_params: 3 x (centres, log_sigmas)
// Output: (8, 960) f32 = [g_features(512) | rbf256 | rbf128 | rbf64]
// =====================================================================
extern "C" void kernel_launch(void* const* d_in, const int* in_sizes, int n_in,
                              void* d_out, int out_size, void* d_ws,
                              size_t ws_size, hipStream_t stream) {
  (void)in_sizes; (void)n_in; (void)out_size; (void)ws_size;

  const float* pc = (const float*)d_in[0];
  const float *gW[6], *gB[6], *gG[6], *gBe[6];
  for (int l = 0; l < 6; ++l) {
    gW[l]  = (const float*)d_in[1 + 4 * l];
    gB[l]  = (const float*)d_in[2 + 4 * l];
    gG[l]  = (const float*)d_in[3 + 4 * l];
    gBe[l] = (const float*)d_in[4 + 4 * l];
  }
  const float *rbfC[3], *rbfS[3];
  for (int i = 0; i < 3; ++i) {
    rbfC[i] = (const float*)d_in[73 + 2 * i];
    rbfS[i] = (const float*)d_in[74 + 2 * i];
  }
  float* out = (float*)d_out;

  // workspace bump allocator (256B aligned)
  uint8_t* wsp = (uint8_t*)d_ws;
  size_t off = 0;
  auto alloc = [&](size_t bytes) -> void* {
    off = (off + 255) & ~(size_t)255;
    void* p = wsp + off;
    off += bytes;
    return p;
  };

  const int dims[7] = {3, 32, 64, 64, 128, 256, 512};
  const int Npts = 4096;

  _Float16* actA = (_Float16*)alloc((size_t)Npts * 512 * sizeof(_Float16));
  _Float16* actB = (_Float16*)alloc((size_t)Npts * 512 * sizeof(_Float16));
  _Float16* Wp[6];
  for (int l = 0; l < 6; ++l) {
    const int K = dims[l];
    const int Kpad = (K + 31) & ~31;
    const int N = dims[l + 1];
    Wp[l] = (_Float16*)alloc((size_t)N * Kpad * sizeof(_Float16));
  }
  int*   fidx = (int*)alloc(1024 * sizeof(int));
  float* xyz1 = (float*)alloc(1024 * 3 * sizeof(float));
  float* xyz2 = (float*)alloc(256 * 3 * sizeof(float));
  float* xyz3 = (float*)alloc(64 * 3 * sizeof(float));
  float* xyz4 = (float*)alloc(16 * 3 * sizeof(float));

  // ---- one-time weight pre-pack (f32 -> padded f16, B-friendly layout) ----
  for (int l = 0; l < 6; ++l) {
    const int K = dims[l];
    const int Kpad = (K + 31) & ~31;
    const int N = dims[l + 1];
    const int tot = N * Kpad;
    pack_w_f16<<<(tot + 255) / 256, 256, 0, stream>>>(gW[l], Wp[l], N, K, Kpad);
  }

  for (int b = 0; b < 8; ++b) {
    const float* pcB = pc + (size_t)b * Npts * 6;
    float* outB = out + (size_t)b * 960;

    // ---- global MLP (WMMA) + mean -> outB[0:512] ----
    build_xyz_f16<<<(Npts + 255) / 256, 256, 0, stream>>>(pcB, actA, Npts, 32);
    _Float16* cur = actA;
    _Float16* nxt = actB;
    for (int l = 0; l < 6; ++l) {
      const int Kpad = (dims[l] + 31) & ~31;
      const int N = dims[l + 1];
      dim3 grid(Npts / 64, N / 16);
      linear_relu_wmma<<<grid, 128, 0, stream>>>(cur, Wp[l], gB[l], gG[l],
                                                 gBe[l], nxt, Kpad, N);
      _Float16* t = cur; cur = nxt; nxt = t;
    }
    colmean_kernel<<<4, 128, 0, stream>>>(cur, Npts, 512, outB);

    // ---- FPS chain (feature MLPs are dead code; only coordinates matter) ----
    fps_kernel<<<1, 512, 0, stream>>>(pcB, Npts, 6, fidx, 1024);
    gather_xyz<<<4, 256, 0, stream>>>(pcB, 6, fidx, xyz1, 1024);
    fps_kernel<<<1, 512, 0, stream>>>(xyz1, 1024, 3, fidx, 256);
    gather_xyz<<<1, 256, 0, stream>>>(xyz1, 3, fidx, xyz2, 256);
    fps_kernel<<<1, 512, 0, stream>>>(xyz2, 256, 3, fidx, 64);
    gather_xyz<<<1, 64, 0, stream>>>(xyz2, 3, fidx, xyz3, 64);
    fps_kernel<<<1, 512, 0, stream>>>(xyz3, 64, 3, fidx, 16);
    gather_xyz<<<1, 16, 0, stream>>>(xyz3, 3, fidx, xyz4, 16);

    // ---- RBF means -> outB[512:960] ----
    rbf_kernel<<<2, 128, 0, stream>>>(xyz2, 256, rbfC[0], rbfS[0], 256,
                                      outB + 512);
    rbf_kernel<<<1, 128, 0, stream>>>(xyz3, 64, rbfC[1], rbfS[1], 128,
                                      outB + 512 + 256);
    rbf_kernel<<<1, 64, 0, stream>>>(xyz4, 16, rbfC[2], rbfS[2], 64,
                                     outB + 512 + 256 + 128);
  }
}
